// FCNet_16037407883621
// MI455X (gfx1250) — compile-verified, matrix operand order checked by reading, not run on hardware
//
#include <hip/hip_runtime.h>
#include <hip/hip_bf16.h>

typedef __attribute__((ext_vector_type(16))) _Float16 v16h;
typedef __attribute__((ext_vector_type(8)))  _Float16 v8h;
typedef __attribute__((ext_vector_type(8)))  float    v8f;

#define DEEP_REG 0.1f
#define BN_EPS   1e-5f

__constant__ int IDX18v[18] = {0,1,2,3,4,5,6,7,8,9,10,11,12,13,14,15,45,46};

// ---------------------------------------------------------------------------
// AQ_Tb[b,p,i] = sum_m AQ[b,m,i] * bv[b,p,m]          (b:2, p:12^3, m:30, i:47)
// ---------------------------------------------------------------------------
__global__ void k_aqtb(const float* __restrict__ bvec, const float* __restrict__ AQ,
                       float* __restrict__ aqtb, int Pb, int M, int C) {
    long idx = (long)blockIdx.x * blockDim.x + threadIdx.x;
    long total = (long)2 * Pb * C;
    if (idx >= total) return;
    int i  = (int)(idx % C);
    long t = idx / C;
    int p  = (int)(t % Pb);
    int bb = (int)(t / Pb);
    const float* aq = AQ + (long)bb * M * C;
    const float* bv = bvec + ((long)bb * Pb + p) * M;
    float s = 0.f;
    for (int m = 0; m < M; ++m) s += aq[(long)m * C + i] * bv[m];
    aqtb[idx] = s;
}

// ---------------------------------------------------------------------------
// Per-batch: G = AQ^T AQ ; invert (G + 0.1 I) [47x47] and
// (G[idx18,idx18] + 0.01 I) [18x18] via Gauss-Jordan in LDS.
// ---------------------------------------------------------------------------
__global__ __launch_bounds__(256) void k_prepare_inv(const float* __restrict__ AQ,
                                                     float* __restrict__ a47inv,
                                                     float* __restrict__ a18inv,
                                                     int M /*30*/) {
    const int bb  = blockIdx.x;
    const int tid = threadIdx.x;
    __shared__ float G[47 * 47];
    __shared__ float AUG[47 * 94];
    __shared__ float col[47];
    const float* aq = AQ + (long)bb * M * 47;

    for (int e = tid; e < 47 * 47; e += 256) {
        int i = e / 47, j = e % 47;
        float s = 0.f;
        for (int m = 0; m < M; ++m) s += aq[m * 47 + i] * aq[m * 47 + j];
        G[e] = s;
    }
    __syncthreads();

    // ---- 47x47 inverse ----
    for (int e = tid; e < 47 * 94; e += 256) {
        int i = e / 94, j = e % 94;
        float v;
        if (j < 47) v = G[i * 47 + j] + ((i == j) ? DEEP_REG : 0.f);
        else        v = ((j - 47) == i) ? 1.f : 0.f;
        AUG[e] = v;
    }
    __syncthreads();
    for (int k = 0; k < 47; ++k) {
        float piv = AUG[k * 94 + k];
        __syncthreads();
        float pinv = 1.f / piv;
        for (int j = tid; j < 94; j += 256) AUG[k * 94 + j] *= pinv;
        __syncthreads();
        for (int i = tid; i < 47; i += 256) col[i] = AUG[i * 94 + k];
        __syncthreads();
        for (int e = tid; e < 47 * 94; e += 256) {
            int i = e / 94, j = e % 94;
            if (i != k) AUG[e] -= col[i] * AUG[k * 94 + j];
        }
        __syncthreads();
    }
    for (int e = tid; e < 47 * 47; e += 256) {
        int i = e / 47, j = e % 47;
        a47inv[(long)bb * 2209 + e] = AUG[i * 94 + 47 + j];
    }
    __syncthreads();

    // ---- 18x18 inverse (reuse AUG as 18x36) ----
    for (int e = tid; e < 18 * 36; e += 256) {
        int i = e / 36, j = e % 36;
        float v;
        if (j < 18) v = G[IDX18v[i] * 47 + IDX18v[j]] + ((i == j) ? 0.01f : 0.f);
        else        v = ((j - 18) == i) ? 1.f : 0.f;
        AUG[e] = v;
    }
    __syncthreads();
    for (int k = 0; k < 18; ++k) {
        float piv = AUG[k * 36 + k];
        __syncthreads();
        float pinv = 1.f / piv;
        for (int j = tid; j < 36; j += 256) AUG[k * 36 + j] *= pinv;
        __syncthreads();
        for (int i = tid; i < 18; i += 256) col[i] = AUG[i * 36 + k];
        __syncthreads();
        for (int e = tid; e < 18 * 36; e += 256) {
            int i = e / 36, j = e % 36;
            if (i != k) AUG[e] -= col[i] * AUG[k * 36 + j];
        }
        __syncthreads();
    }
    for (int e = tid; e < 18 * 18; e += 256) {
        int i = e / 18, j = e % 18;
        a18inv[(long)bb * 324 + e] = AUG[i * 36 + 18 + j];
    }
}

// ---------------------------------------------------------------------------
// c0: scatter(A18inv @ AQ_Tb[...,idx18]) -> c[p][47].  One block per point.
// ---------------------------------------------------------------------------
__global__ __launch_bounds__(64) void k_c0(const float* __restrict__ aqtb,
                                           const float* __restrict__ a18inv,
                                           float* __restrict__ c, int Pb) {
    int p  = blockIdx.x;
    int bb = p / Pb;
    __shared__ float rhs[18];
    const float* at = aqtb + (long)p * 47;
    if (threadIdx.x < 18) rhs[threadIdx.x] = at[IDX18v[threadIdx.x]];
    __syncthreads();
    int j = threadIdx.x;
    if (j < 47) {
        int jh = -1;
        if (j < 16) jh = j; else if (j == 45) jh = 16; else if (j == 46) jh = 17;
        float v = 0.f;
        if (jh >= 0) {
            const float* inv = a18inv + (long)bb * 324 + jh * 18;
            for (int i = 0; i < 18; ++i) v += inv[i] * rhs[i];
        }
        c[(long)p * 47 + j] = v;
    }
}

// ---------------------------------------------------------------------------
// Convert c [P][47] f32 -> padded NDHWC f16 activation [P][64]
// ---------------------------------------------------------------------------
__global__ void k_c_to_act(const float* __restrict__ c, _Float16* __restrict__ act, long P) {
    long idx = (long)blockIdx.x * blockDim.x + threadIdx.x;
    long total = P * 64;
    if (idx >= total) return;
    int ch = (int)(idx % 64);
    long p = idx / 64;
    float v = (ch < 47) ? c[p * 47 + ch] : 0.f;
    act[idx] = (_Float16)v;
}

// ---------------------------------------------------------------------------
// Weight repack: f32 OIDHW [Cout][Cin][T] -> f16 in WMMA B-FRAGMENT layout.
// Record index: (((t*KCH + kc)*Coutp + col)*2 + half) -> 16 contiguous halves:
//   halves 0..7  = K = kc*32 + half*8 + i
//   halves 8..15 = K = kc*32 + half*8 + 16 + (i-8)
// Each lane's B fragment becomes two contiguous 16B loads.
// ---------------------------------------------------------------------------
__global__ void k_repack_w(const float* __restrict__ w, _Float16* __restrict__ wq,
                           int Cout, int Cin, int Cinp, int Coutp, int T) {
    long idx = (long)blockIdx.x * blockDim.x + threadIdx.x;
    long total = (long)T * Cinp * Coutp;          // same element count as dense
    if (idx >= total) return;
    const int KCH = Cinp >> 5;
    int  i    = (int)(idx & 15);
    long r    = idx >> 4;
    int  half = (int)(r & 1);  r >>= 1;
    int  col  = (int)(r % Coutp);
    long r2   = r / Coutp;
    int  kc   = (int)(r2 % KCH);
    int  t    = (int)(r2 / KCH);
    int  koff = half * 8 + ((i < 8) ? i : (8 + i));  // {0..7, 16..23} + half*8
    int  K    = kc * 32 + koff;
    float v = (col < Cout && K < Cin) ? w[((long)col * Cin + K) * T + t] : 0.f;
    wq[idx] = (_Float16)v;
}

// ---------------------------------------------------------------------------
// Implicit-GEMM 3D conv via v_wmma_f32_16x16x32_f16.
// One wave32 computes a 16(M=positions) x 32(N=out channels) tile using TWO
// accumulators that share each A fragment.  K loop fully unrolled (KCH is a
// template constant) -> static chain of v_wmma ops.  A and B fragments are
// each two contiguous 16B vector loads (B is pre-swizzled into fragment
// layout by k_repack_w).
// MODE: 0 = 3x3x3 SAME, 1 = 3x3x3 VALID, 2 = 1x1x1
// Axis mapping (matches reference transposes): kw->x, kd->y, kh->z offsets.
// ---------------------------------------------------------------------------
template <int KCH, int MODE>
__global__ __launch_bounds__(32) void k_conv_wmma(
    const _Float16* __restrict__ act, const _Float16* __restrict__ wq,
    const float* __restrict__ bias, float* __restrict__ y,
    int Sin, int Sout, int Cout, int Coutp)
{
    constexpr int Cinp = KCH * 32;
    constexpr int T    = (MODE == 2) ? 1 : 27;

    const int lane = threadIdx.x & 31;
    const int row  = lane & 15;
    const int half = lane >> 4;
    const int m0 = blockIdx.x * 16;
    const int n0 = blockIdx.y * 32;

    // decompose this lane's output position (A-matrix row)
    int m  = m0 + row;
    int oz = m % Sout; int tmp = m / Sout;
    int oy = tmp % Sout; tmp /= Sout;
    int ox = tmp % Sout; int bb = tmp / Sout;

    const int col0 = n0 + row;              // B-matrix columns (out channels)
    const int col1 = col0 + 16;

    v8f acc0 = {};
    v8f acc1 = {};
    for (int t = 0; t < T; ++t) {
        int kd = 0, kh = 0, kw = 0;
        if (MODE != 2) { kd = t / 9; kh = (t / 3) % 3; kw = t % 3; }
        int ix = ox + kw, iy = oy + kd, iz = oz + kh;
        if (MODE == 0) { ix -= 1; iy -= 1; iz -= 1; }
        bool valid = (unsigned)ix < (unsigned)Sin && (unsigned)iy < (unsigned)Sin &&
                     (unsigned)iz < (unsigned)Sin;
        const long base = ((((long)bb * Sin + ix) * Sin + iy) * Sin + iz) * Cinp;

        // prefetch this lane's activation row for the NEXT tap (gfx1250 path)
        if (MODE != 2 && t + 1 < T) {
            int t2 = t + 1;
            int kd2 = t2 / 9, kh2 = (t2 / 3) % 3, kw2 = t2 % 3;
            int ix2 = ox + kw2, iy2 = oy + kd2, iz2 = oz + kh2;
            if (MODE == 0) { ix2 -= 1; iy2 -= 1; iz2 -= 1; }
            if ((unsigned)ix2 < (unsigned)Sin && (unsigned)iy2 < (unsigned)Sin &&
                (unsigned)iz2 < (unsigned)Sin) {
                const long b2 = ((((long)bb * Sin + ix2) * Sin + iy2) * Sin + iz2) * Cinp;
                __builtin_prefetch(act + b2, 0, 3);
            }
        }

        #pragma unroll
        for (int kc = 0; kc < KCH; ++kc) {
            const int k0 = kc * 32 + half * 8;    // lane's first K of this chunk
            v16h a;
            if (valid) {
                const v8h* ap = (const v8h*)(act + base + k0);
                v8h lo = ap[0];
                v8h hi = ap[2];                   // +16 halves
                #pragma unroll
                for (int i = 0; i < 8; ++i) { a[i] = lo[i]; a[8 + i] = hi[i]; }
            } else {
                #pragma unroll
                for (int i = 0; i < 16; ++i) a[i] = (_Float16)0.f;
            }
            // B fragments: pre-swizzled 32-byte records -> 2 x b128 each
            const long rec0 = ((((long)t * KCH + kc) * Coutp + col0) * 2 + half) * 16;
            const long rec1 = ((((long)t * KCH + kc) * Coutp + col1) * 2 + half) * 16;
            const v8h* bp0 = (const v8h*)(wq + rec0);
            const v8h* bp1 = (const v8h*)(wq + rec1);
            v8h b0l = bp0[0], b0h = bp0[1];
            v8h b1l = bp1[0], b1h = bp1[1];
            v16h b0, b1;
            #pragma unroll
            for (int i = 0; i < 8; ++i) {
                b0[i] = b0l[i]; b0[8 + i] = b0h[i];
                b1[i] = b1l[i]; b1[8 + i] = b1h[i];
            }
            acc0 = __builtin_amdgcn_wmma_f32_16x16x32_f16(
                false, a, false, b0, (short)0, acc0, false, false);
            acc1 = __builtin_amdgcn_wmma_f32_16x16x32_f16(
                false, a, false, b1, (short)0, acc1, false, false);
        }
    }

    float bv0 = (col0 < Cout) ? bias[col0] : 0.f;
    float bv1 = (col1 < Cout) ? bias[col1] : 0.f;
    #pragma unroll
    for (int r = 0; r < 8; ++r) {
        int mm = m0 + r + half * 8;               // C/D layout: VGPR r holds M=r (+8 hi half)
        y[(long)mm * Coutp + col0] = acc0[r] + bv0;
        y[(long)mm * Coutp + col1] = acc1[r] + bv1;
    }
}

// ---------------------------------------------------------------------------
// Per-channel mean / rstd over all positions (batch+spatial), one block/channel
// ---------------------------------------------------------------------------
__global__ __launch_bounds__(256) void k_bn_stats(const float* __restrict__ y,
                                                  float* __restrict__ mean,
                                                  float* __restrict__ rstd,
                                                  int P, int Coutp) {
    int c = blockIdx.x;
    __shared__ float s1[256], s2[256];
    float a = 0.f, b = 0.f;
    for (int p = threadIdx.x; p < P; p += 256) {
        float v = y[(long)p * Coutp + c];
        a += v; b += v * v;
    }
    s1[threadIdx.x] = a; s2[threadIdx.x] = b;
    __syncthreads();
    for (int s = 128; s > 0; s >>= 1) {
        if (threadIdx.x < s) { s1[threadIdx.x] += s1[threadIdx.x + s];
                               s2[threadIdx.x] += s2[threadIdx.x + s]; }
        __syncthreads();
    }
    if (threadIdx.x == 0) {
        float mu  = s1[0] / (float)P;
        float var = s2[0] / (float)P - mu * mu;
        mean[c] = mu;
        rstd[c] = rsqrtf(var + BN_EPS);
    }
}

// ---------------------------------------------------------------------------
// Fused (optional BN) + ReLU + f32 -> padded f16 activation
// ---------------------------------------------------------------------------
__global__ void k_bn_act(const float* __restrict__ y, const float* __restrict__ mean,
                         const float* __restrict__ rstd, const float* __restrict__ gamma,
                         const float* __restrict__ beta, _Float16* __restrict__ act,
                         long P, int Cout, int Coutp, int do_bn) {
    long idx = (long)blockIdx.x * blockDim.x + threadIdx.x;
    long total = P * Coutp;
    if (idx >= total) return;
    int c = (int)(idx % Coutp);
    float v = 0.f;
    if (c < Cout) {
        v = y[idx];
        if (do_bn) v = gamma[c] * (v - mean[c]) * rstd[c] + beta[c];
        v = fmaxf(v, 0.f);
    }
    act[idx] = (_Float16)v;
}

// ---------------------------------------------------------------------------
// Fused gate + deep-correction solve:
//  csd = y[:47]*sigmoid(y[47:]) (+ center-crop of previous c)
//  rhs = AQ_Tb[crop n] + 0.1*csd ;  c_new = A47inv @ rhs
// ---------------------------------------------------------------------------
__global__ __launch_bounds__(64) void k_dc(const float* __restrict__ y,
                                           const float* __restrict__ aqtb,
                                           const float* __restrict__ cprev,
                                           const float* __restrict__ a47inv,
                                           float* __restrict__ cnew,
                                           int Sfull, int Sout, int Sprev,
                                           int n, int use_prev) {
    int p  = blockIdx.x;
    int Pb = Sout * Sout * Sout;
    int bb = p / Pb;
    int r  = p % Pb;
    int oz = r % Sout; int t = r / Sout;
    int oy = t % Sout; int ox = t / Sout;
    __shared__ float rhs[47];
    int tid = threadIdx.x;
    if (tid < 47) {
        float g = y[(long)p * 96 + tid];
        float h = y[(long)p * 96 + 47 + tid];
        float csd = g * (1.f / (1.f + __expf(-h)));
        if (use_prev) {
            long pp = (((long)bb * Sprev + (ox + 1)) * Sprev + (oy + 1)) * Sprev + (oz + 1);
            csd += cprev[pp * 47 + tid];
        }
        long pa = (((long)bb * Sfull + (ox + n)) * Sfull + (oy + n)) * Sfull + (oz + n);
        rhs[tid] = aqtb[pa * 47 + tid] + DEEP_REG * csd;
    }
    __syncthreads();
    if (tid < 47) {
        const float* inv = a47inv + (long)bb * 2209 + tid * 47;
        float v = 0.f;
        for (int i = 0; i < 47; ++i) v += inv[i] * rhs[i];
        cnew[(long)p * 47 + tid] = v;
    }
}

// ---------------------------------------------------------------------------
// Host-side conv dispatch over the 9 fixed layer shapes
// ---------------------------------------------------------------------------
static void launch_conv(int layer, const _Float16* a, const _Float16* w,
                        const float* bias, float* y, int Sin, int Sout,
                        int Cout, int Coutp, long Pout, hipStream_t stream) {
    dim3 grid((unsigned)(Pout / 16), (unsigned)(Coutp / 32));
    dim3 blk(32);
    switch (layer) {
        case 0: k_conv_wmma<2, 0><<<grid, blk, 0, stream>>>(a, w, bias, y, Sin, Sout, Cout, Coutp); break;
        case 1: k_conv_wmma<2, 0><<<grid, blk, 0, stream>>>(a, w, bias, y, Sin, Sout, Cout, Coutp); break;
        case 2: k_conv_wmma<4, 0><<<grid, blk, 0, stream>>>(a, w, bias, y, Sin, Sout, Cout, Coutp); break;
        case 3: k_conv_wmma<6, 0><<<grid, blk, 0, stream>>>(a, w, bias, y, Sin, Sout, Cout, Coutp); break;
        case 4: k_conv_wmma<8, 0><<<grid, blk, 0, stream>>>(a, w, bias, y, Sin, Sout, Cout, Coutp); break;
        case 5: k_conv_wmma<10, 0><<<grid, blk, 0, stream>>>(a, w, bias, y, Sin, Sout, Cout, Coutp); break;
        case 6: k_conv_wmma<12, 0><<<grid, blk, 0, stream>>>(a, w, bias, y, Sin, Sout, Cout, Coutp); break;
        case 7: k_conv_wmma<14, 1><<<grid, blk, 0, stream>>>(a, w, bias, y, Sin, Sout, Cout, Coutp); break;
        case 8: k_conv_wmma<16, 2><<<grid, blk, 0, stream>>>(a, w, bias, y, Sin, Sout, Cout, Coutp); break;
    }
}

// ---------------------------------------------------------------------------
// Host orchestration
// ---------------------------------------------------------------------------
extern "C" void kernel_launch(void* const* d_in, const int* in_sizes, int n_in,
                              void* d_out, int out_size, void* d_ws, size_t ws_size,
                              hipStream_t stream) {
    (void)in_sizes; (void)n_in; (void)out_size; (void)ws_size;

    const float* d_b  = (const float*)d_in[0];   // (2,12,12,12,30,1)
    const float* d_AQ = (const float*)d_in[1];   // (2,30,47)

    // channel tables
    const int ins[9]   = {47, 64, 128, 192, 256, 320, 384, 448, 512};
    const int outs[9]  = {64, 128, 192, 256, 320, 384, 448, 512, 94};
    const int cinp[9]  = {64, 64, 128, 192, 256, 320, 384, 448, 512};
    const int coutp[9] = {64, 128, 192, 256, 320, 384, 448, 512, 96};

    // workspace layout
    char* ws = (char*)d_ws;
    size_t off = 0;
    auto take = [&](size_t bytes) -> char* {
        char* p = ws + off;
        off = (off + bytes + 255) & ~(size_t)255;
        return p;
    };
    float*    aqtb   = (float*)take((size_t)2 * 1728 * 47 * 4);
    float*    a47inv = (float*)take((size_t)2 * 2209 * 4);
    float*    a18inv = (float*)take((size_t)2 * 324 * 4);
    float*    c_a    = (float*)take((size_t)2 * 1728 * 47 * 4);
    float*    c_b    = (float*)take((size_t)2 * 1728 * 47 * 4);
    _Float16* act_a  = (_Float16*)take((size_t)3456 * 512 * 2);
    _Float16* act_b  = (_Float16*)take((size_t)3456 * 512 * 2);
    float*    ybuf   = (float*)take((size_t)3456 * 512 * 4);
    _Float16* wqbuf  = (_Float16*)take((size_t)27 * 512 * 512 * 2);
    float*    meanb  = (float*)take((size_t)512 * 4);
    float*    rstdb  = (float*)take((size_t)512 * 4);

    // ---- small-solve preparation ----
    {
        long total = (long)2 * 1728 * 47;
        k_aqtb<<<dim3((unsigned)((total + 255) / 256)), dim3(256), 0, stream>>>(
            d_b, d_AQ, aqtb, 1728, 30, 47);
        k_prepare_inv<<<dim3(2), dim3(256), 0, stream>>>(d_AQ, a47inv, a18inv, 30);
        k_c0<<<dim3(2 * 1728), dim3(64), 0, stream>>>(aqtb, a18inv, c_a, 1728);
    }

    float* c_cur = c_a;
    float* c_nxt = c_b;

    for (int ci = 0; ci < 4; ++ci) {
        const int Sin0 = 12 - 2 * ci;                     // cascade input spatial
        const int base = 2 + ci * 32;                     // param array base index

        // convert c (f32, 47ch) -> padded f16 NDHWC activation
        {
            long P = (long)2 * Sin0 * Sin0 * Sin0;
            long total = P * 64;
            k_c_to_act<<<dim3((unsigned)((total + 255) / 256)), dim3(256), 0, stream>>>(
                c_cur, act_a, P);
        }

        _Float16* a_in  = act_a;
        _Float16* a_out = act_b;

        for (int i = 0; i < 9; ++i) {
            const int Sl_in = (i == 8) ? (Sin0 - 2) : Sin0;
            const int Sl_out= (i < 7) ? Sin0 : (Sin0 - 2);
            const int T     = (i == 8) ? 1 : 27;
            const int Cin   = ins[i],  Cip = cinp[i];
            const int Cout  = outs[i], Cop = coutp[i];
            const long Pout = (long)2 * Sl_out * Sl_out * Sl_out;

            const float* w_p = (const float*)d_in[base + i];
            const float* b_p = (const float*)d_in[base + 9 + i];

            // repack weights -> f16 B-fragment records [T][KCH][Cop][2][16]
            {
                long total = (long)T * Cip * Cop;
                k_repack_w<<<dim3((unsigned)((total + 255) / 256)), dim3(256), 0, stream>>>(
                    w_p, wqbuf, Cout, Cin, Cip, Cop, T);
            }
            // WMMA conv
            launch_conv(i, a_in, wqbuf, b_p, ybuf, Sl_in, Sl_out, Cout, Cop, Pout, stream);

            if (i < 8) {
                const int do_bn = (i < 7) ? 1 : 0;
                const float* g_p = do_bn ? (const float*)d_in[base + 18 + i] : (const float*)d_in[base + 9];
                const float* be_p= do_bn ? (const float*)d_in[base + 25 + i] : (const float*)d_in[base + 9];
                if (do_bn)
                    k_bn_stats<<<dim3(Cout), dim3(256), 0, stream>>>(ybuf, meanb, rstdb,
                                                                     (int)Pout, Cop);
                long total = Pout * Cop;
                k_bn_act<<<dim3((unsigned)((total + 255) / 256)), dim3(256), 0, stream>>>(
                    ybuf, meanb, rstdb, g_p, be_p, a_out, Pout, Cout, Cop, do_bn);
                _Float16* t = a_in; a_in = a_out; a_out = t;
            }
        }

        // fused gate + solve
        {
            const int Sc = Sin0 - 2;                      // new c spatial
            const int Pc = 2 * Sc * Sc * Sc;
            float* dst = (ci == 3) ? (float*)d_out : c_nxt;
            k_dc<<<dim3(Pc), dim3(64), 0, stream>>>(
                ybuf, aqtb, c_cur, a47inv, dst,
                12, Sc, Sin0, ci + 1, (ci > 0) ? 1 : 0);
            float* t = c_cur; c_cur = c_nxt; c_nxt = t;
        }
    }
}
